// SparseMoEBlockFast_12841952215338
// MI455X (gfx1250) — matrix operand order repaired and checked by reading, not compiled, not run on hardware
//
#include <hip/hip_runtime.h>
#include <stdint.h>

// ---- problem constants (from reference) ----
#define NUM_EXPERTS 8
#define HIDDEN 1024
#define INTER  2816
#define TOKENS 8192          // 4 * 2048
#define MTILE  128           // token tile
#define NTILE  64            // output-col tile
#define KTILE  64            // K stage in LDS
#define KPAD   8             // LDS row padding (halves)
#define MTILES (TOKENS / MTILE)   // 64

#if defined(__AMDGCN__) && __has_builtin(__builtin_amdgcn_tensor_load_to_lds)
#define HAVE_TDM 1
#else
#define HAVE_TDM 0
#endif

typedef __attribute__((ext_vector_type(16))) __bf16 v16bf;
typedef __attribute__((ext_vector_type(8)))  float  v8f;

union FragB {
  uint4  q[2];
  v16bf  v;
};

static __device__ __forceinline__ unsigned short f2bf(float f) {
  union { float f; uint32_t u; } x; x.f = f;
  uint32_t u = x.u;
  uint32_t r = (u + 0x7FFFu + ((u >> 16) & 1u)) >> 16;  // round-to-nearest-even
  return (unsigned short)r;
}

#if HAVE_TDM
typedef __attribute__((ext_vector_type(4))) unsigned int v4u;
typedef __attribute__((ext_vector_type(8))) int          v8i;
typedef __attribute__((ext_vector_type(4))) int          v4i;

// DMA one 128-row x 64-col bf16 tile (row stride = row_len elements) from
// global into LDS, with TDM hardware padding of 4 dwords after every 32
// dwords -> matches the [row][KTILE + KPAD] LDS layout exactly.
// 6-arg builtin form: (g0, g1, g2, g3, extra_group, cpol)
static __device__ __forceinline__ void tdm_load_a_tile(const unsigned short* gsrc,
                                                       unsigned lds_addr,
                                                       unsigned row_len) {
  uint64_t ga = (uint64_t)(uintptr_t)gsrc;
  v4u g0;
  g0.x = 1u;                                   // count = 1 valid descriptor
  g0.y = lds_addr;                             // LDS byte address (tile start)
  g0.z = (unsigned)ga;                         // global_addr[31:0]
  g0.w = (unsigned)((ga >> 32) & 0x1FFFFFFu)   // global_addr[56:32]
       | (2u << 30);                           // type = 2 ("image")
  v8i g1;
  // data_size=1 (2B), pad_enable, pad_interval=4 (32 dwords), pad_amount=3 (4 dwords)
  g1[0] = (int)((1u << 16) | (1u << 20) | (4u << 22) | (3u << 25));
  g1[1] = (int)(row_len << 16);                              // tensor_dim0[15:0]
  g1[2] = (int)((row_len >> 16) | ((unsigned)TOKENS << 16)); // dim0 hi | tensor_dim1 lo
  g1[3] = (int)(64u << 16);                                  // dim1 hi=0 | tile_dim0=64
  g1[4] = (int)128;                                          // tile_dim1=128, tile_dim2=0
  g1[5] = (int)row_len;                                      // tensor_dim0_stride lo
  g1[6] = 0;                                                 // stride hi | dim1_stride lo
  g1[7] = 0;
  v4i z4; z4[0] = z4[1] = z4[2] = z4[3] = 0;                 // 2D tile: groups 2/3 unused
  v8i z8;
#pragma unroll
  for (int i = 0; i < 8; ++i) z8[i] = 0;
  __builtin_amdgcn_tensor_load_to_lds(g0, g1, z4, z4, z8, 0);
}
#endif

// ---------------------------------------------------------------------------
// init: zero output accumulator and tile-activity flags
// ---------------------------------------------------------------------------
__global__ void moe_init_kernel(float* __restrict__ out, int* __restrict__ flags,
                                int n_out, int n_flags) {
  int i = blockIdx.x * blockDim.x + threadIdx.x;
  int stride = gridDim.x * blockDim.x;
  for (int j = i; j < n_out; j += stride) out[j] = 0.f;
  if (i < n_flags) flags[i] = 0;
}

// ---------------------------------------------------------------------------
// router: logits -> softmax -> top2 -> renormalized combine weights.
// Also converts x (fp32) to bf16 for the WMMA A operand, and marks which
// (expert, 128-token tile) pairs are active.
// ---------------------------------------------------------------------------
__global__ __launch_bounds__(256) void moe_router_kernel(
    const float* __restrict__ x, const float* __restrict__ rw,
    unsigned short* __restrict__ xb, float* __restrict__ cw,
    int* __restrict__ flags) {
  __shared__ float rws[HIDDEN * NUM_EXPERTS];  // 32 KB
  const int tid = threadIdx.x;
  for (int i = tid; i < HIDDEN * NUM_EXPERTS; i += 256) rws[i] = rw[i];
  __syncthreads();

  const int t = blockIdx.x * 256 + tid;
  float acc[NUM_EXPERTS];
#pragma unroll
  for (int e = 0; e < NUM_EXPERTS; ++e) acc[e] = 0.f;

  const size_t base = (size_t)t * HIDDEN;
  for (int h = 0; h < HIDDEN; ++h) {
    float v = x[base + h];
    xb[base + h] = f2bf(v);
#pragma unroll
    for (int e = 0; e < NUM_EXPERTS; ++e) acc[e] += v * rws[h * NUM_EXPERTS + e];
  }

  float mx = acc[0];
#pragma unroll
  for (int e = 1; e < NUM_EXPERTS; ++e) mx = fmaxf(mx, acc[e]);
  float p[NUM_EXPERTS];
#pragma unroll
  for (int e = 0; e < NUM_EXPERTS; ++e) p[e] = __expf(acc[e] - mx);

  int i1 = 0; float v1 = p[0];
#pragma unroll
  for (int e = 1; e < NUM_EXPERTS; ++e) { if (p[e] > v1) { v1 = p[e]; i1 = e; } }
  int i2 = -1; float v2 = -1.f;
#pragma unroll
  for (int e = 0; e < NUM_EXPERTS; ++e) {
    if (e != i1 && p[e] > v2) { v2 = p[e]; i2 = e; }
  }
  const float inv = 1.f / (v1 + v2);
  const float w1 = v1 * inv, w2 = v2 * inv;
#pragma unroll
  for (int e = 0; e < NUM_EXPERTS; ++e)
    cw[(size_t)t * NUM_EXPERTS + e] = (e == i1) ? w1 : ((e == i2) ? w2 : 0.f);

  const int tile = t >> 7;
  atomicOr(&flags[i1 * MTILES + tile], 1);
  atomicOr(&flags[i2 * MTILES + tile], 1);
}

// ---------------------------------------------------------------------------
// gate/up GEMM + SwiGLU: inter[t, n] = silu(x@Wg)[t,n] * (x@Wu)[t,n]  (bf16)
// M = TOKENS, N = INTER, K = HIDDEN. Dual-B WMMA bf16, double-buffered LDS,
// A tile via TDM (tensor_load_to_lds), B tiles via load->cvt->LDS.
// ---------------------------------------------------------------------------
__global__ __launch_bounds__(256) void moe_gateup_kernel(
    const unsigned short* __restrict__ xb,
    const float* __restrict__ wg, const float* __restrict__ wu,
    unsigned short* __restrict__ inter, const int* __restrict__ flags) {
  const int mblk = blockIdx.y;
  if (flags[mblk] == 0) return;  // no token in this tile routed here
  const int n0 = blockIdx.x * NTILE;
  const int t0 = mblk * MTILE;

  __shared__ unsigned short As[2][MTILE][KTILE + KPAD];   // 2 x 18 KB
  __shared__ unsigned short Bgs[2][NTILE][KTILE + KPAD];  // 2 x  9 KB
  __shared__ unsigned short Bus[2][NTILE][KTILE + KPAD];  // 2 x  9 KB

  const int tid  = threadIdx.x;
  const int lane = tid & 31;
  const int wave = tid >> 5;
  const int wm0  = (wave & 3) * 32;
  const int wn0  = (wave >> 2) * 32;
  const int lr   = lane & 15;
  const int hw   = lane >> 4;

  v8f accg[2][2] = {};
  v8f accu[2][2] = {};
  float rg[16], ru[16];

  auto issue_a = [&](int buf, int k0) {
#if HAVE_TDM
    if (wave == 0)
      tdm_load_a_tile(xb + (size_t)t0 * HIDDEN + k0,
                      (unsigned)(uintptr_t)&As[buf][0][0], HIDDEN);
#else
#pragma unroll
    for (int it = 0; it < 4; ++it) {
      int slot = it * 256 + tid;
      int r = slot >> 3;
      int c = (slot & 7) * 8;
      *(uint4*)&As[buf][r][c] =
          *(const uint4*)(xb + (size_t)(t0 + r) * HIDDEN + k0 + c);
    }
#endif
  };
  auto wait_a = [&]() {
#if HAVE_TDM
    if (wave == 0) __builtin_amdgcn_s_wait_tensorcnt(0);
#endif
  };
  auto load_b_regs = [&](int k0) {
#pragma unroll
    for (int it = 0; it < 16; ++it) {
      int slot = it * 256 + tid;
      int k = slot >> 6;
      int n = slot & 63;
      size_t gidx = (size_t)(k0 + k) * INTER + n0 + n;
      rg[it] = wg[gidx];
      ru[it] = wu[gidx];
    }
    if (k0 + 2 * KTILE < HIDDEN) {  // prefetch two stages ahead
      __builtin_prefetch(wg + (size_t)(k0 + 2 * KTILE) * INTER + n0 + (tid & 63), 0, 1);
      __builtin_prefetch(wu + (size_t)(k0 + 2 * KTILE) * INTER + n0 + (tid & 63), 0, 1);
    }
  };
  auto store_b_lds = [&](int buf) {
#pragma unroll
    for (int it = 0; it < 16; ++it) {
      int slot = it * 256 + tid;
      int k = slot >> 6;
      int n = slot & 63;
      Bgs[buf][n][k] = f2bf(rg[it]);
      Bus[buf][n][k] = f2bf(ru[it]);
    }
  };
  auto compute = [&](int buf) {
#pragma unroll
    for (int ks = 0; ks < KTILE; ks += 32) {
      FragB a[2], bg[2], bu[2];
#pragma unroll
      for (int mi = 0; mi < 2; ++mi) {
        const unsigned short* p = &As[buf][wm0 + mi * 16 + lr][ks + hw * 8];
        a[mi].q[0] = *(const uint4*)p;         // K[kb .. kb+7]
        a[mi].q[1] = *(const uint4*)(p + 16);  // K[kb+16 .. kb+23]
      }
#pragma unroll
      for (int ni = 0; ni < 2; ++ni) {
        const unsigned short* pg = &Bgs[buf][wn0 + ni * 16 + lr][ks + hw * 8];
        bg[ni].q[0] = *(const uint4*)pg;
        bg[ni].q[1] = *(const uint4*)(pg + 16);
        const unsigned short* pu = &Bus[buf][wn0 + ni * 16 + lr][ks + hw * 8];
        bu[ni].q[0] = *(const uint4*)pu;
        bu[ni].q[1] = *(const uint4*)(pu + 16);
      }
#pragma unroll
      for (int mi = 0; mi < 2; ++mi)
#pragma unroll
        for (int ni = 0; ni < 2; ++ni) {
          accg[mi][ni] = __builtin_amdgcn_wmma_f32_16x16x32_bf16(
              false, a[mi].v, false, bg[ni].v, (short)0, accg[mi][ni], false, false);
          accu[mi][ni] = __builtin_amdgcn_wmma_f32_16x16x32_bf16(
              false, a[mi].v, false, bu[ni].v, (short)0, accu[mi][ni], false, false);
        }
    }
  };

  // software pipeline: stage k+1 streaming overlaps WMMA on stage k
  issue_a(0, 0);
  load_b_regs(0);
  store_b_lds(0);
  wait_a();
  __syncthreads();

  int buf = 0;
  for (int k0 = 0; k0 < HIDDEN; k0 += KTILE) {
    const int nk0 = k0 + KTILE;
    const bool has_next = nk0 < HIDDEN;
    if (has_next) {
      issue_a(buf ^ 1, nk0);   // async TDM into other buffer
      load_b_regs(nk0);        // global fp32 loads in flight during compute
    }
    compute(buf);
    if (has_next) {
      store_b_lds(buf ^ 1);
      wait_a();
    }
    __syncthreads();
    buf ^= 1;
  }

  // SwiGLU epilogue -> bf16 inter (fast rcp instead of IEEE divide)
#pragma unroll
  for (int mi = 0; mi < 2; ++mi)
#pragma unroll
    for (int ni = 0; ni < 2; ++ni) {
      const int col = n0 + wn0 + ni * 16 + lr;
#pragma unroll
      for (int v = 0; v < 8; ++v) {
        const int row = t0 + wm0 + mi * 16 + hw * 8 + v;
        float g = accg[mi][ni][v];
        float u = accu[mi][ni][v];
        float s = g * __builtin_amdgcn_rcpf(1.f + __expf(-g));  // silu
        inter[(size_t)row * INTER + col] = f2bf(s * u);
      }
    }
}

// ---------------------------------------------------------------------------
// down GEMM + weighted combine: out[t, n] += cw[t, e] * (inter @ Wd)[t, n]
// M = TOKENS, N = HIDDEN, K = INTER. Experts are launched sequentially,
// so plain read-modify-write on out is race-free.
// ---------------------------------------------------------------------------
__global__ __launch_bounds__(256) void moe_down_kernel(
    const unsigned short* __restrict__ inter, const float* __restrict__ wd,
    const float* __restrict__ cw, int e, float* __restrict__ out,
    const int* __restrict__ flags) {
  const int mblk = blockIdx.y;
  if (flags[mblk] == 0) return;
  const int n0 = blockIdx.x * NTILE;
  const int t0 = mblk * MTILE;

  __shared__ unsigned short As[2][MTILE][KTILE + KPAD];
  __shared__ unsigned short Bs[2][NTILE][KTILE + KPAD];
  __shared__ float cws[MTILE];

  const int tid  = threadIdx.x;
  const int lane = tid & 31;
  const int wave = tid >> 5;
  const int wm0  = (wave & 3) * 32;
  const int wn0  = (wave >> 2) * 32;
  const int lr   = lane & 15;
  const int hw   = lane >> 4;

  if (tid < MTILE) cws[tid] = cw[(size_t)(t0 + tid) * NUM_EXPERTS + e];

  v8f acc[2][2] = {};
  float rb[16];

  auto issue_a = [&](int buf, int k0) {
#if HAVE_TDM
    if (wave == 0)
      tdm_load_a_tile(inter + (size_t)t0 * INTER + k0,
                      (unsigned)(uintptr_t)&As[buf][0][0], INTER);
#else
#pragma unroll
    for (int it = 0; it < 4; ++it) {
      int slot = it * 256 + tid;
      int r = slot >> 3;
      int c = (slot & 7) * 8;
      *(uint4*)&As[buf][r][c] =
          *(const uint4*)(inter + (size_t)(t0 + r) * INTER + k0 + c);
    }
#endif
  };
  auto wait_a = [&]() {
#if HAVE_TDM
    if (wave == 0) __builtin_amdgcn_s_wait_tensorcnt(0);
#endif
  };
  auto load_b_regs = [&](int k0) {
#pragma unroll
    for (int it = 0; it < 16; ++it) {
      int slot = it * 256 + tid;
      int k = slot >> 6;
      int n = slot & 63;
      rb[it] = wd[(size_t)(k0 + k) * HIDDEN + n0 + n];
    }
    if (k0 + 2 * KTILE < INTER) {
      __builtin_prefetch(wd + (size_t)(k0 + 2 * KTILE) * HIDDEN + n0 + (tid & 63), 0, 1);
    }
  };
  auto store_b_lds = [&](int buf) {
#pragma unroll
    for (int it = 0; it < 16; ++it) {
      int slot = it * 256 + tid;
      int k = slot >> 6;
      int n = slot & 63;
      Bs[buf][n][k] = f2bf(rb[it]);
    }
  };
  auto compute = [&](int buf) {
#pragma unroll
    for (int ks = 0; ks < KTILE; ks += 32) {
      FragB a[2], b[2];
#pragma unroll
      for (int mi = 0; mi < 2; ++mi) {
        const unsigned short* p = &As[buf][wm0 + mi * 16 + lr][ks + hw * 8];
        a[mi].q[0] = *(const uint4*)p;
        a[mi].q[1] = *(const uint4*)(p + 16);
      }
#pragma unroll
      for (int ni = 0; ni < 2; ++ni) {
        const unsigned short* pb = &Bs[buf][wn0 + ni * 16 + lr][ks + hw * 8];
        b[ni].q[0] = *(const uint4*)pb;
        b[ni].q[1] = *(const uint4*)(pb + 16);
      }
#pragma unroll
      for (int mi = 0; mi < 2; ++mi)
#pragma unroll
        for (int ni = 0; ni < 2; ++ni)
          acc[mi][ni] = __builtin_amdgcn_wmma_f32_16x16x32_bf16(
              false, a[mi].v, false, b[ni].v, (short)0, acc[mi][ni], false, false);
    }
  };

  issue_a(0, 0);
  load_b_regs(0);
  store_b_lds(0);
  wait_a();
  __syncthreads();

  int buf = 0;
  for (int k0 = 0; k0 < INTER; k0 += KTILE) {
    const int nk0 = k0 + KTILE;
    const bool has_next = nk0 < INTER;
    if (has_next) {
      issue_a(buf ^ 1, nk0);
      load_b_regs(nk0);
    }
    compute(buf);
    if (has_next) {
      store_b_lds(buf ^ 1);
      wait_a();
    }
    __syncthreads();
    buf ^= 1;
  }

#pragma unroll
  for (int mi = 0; mi < 2; ++mi)
#pragma unroll
    for (int ni = 0; ni < 2; ++ni) {
      const int col = n0 + wn0 + ni * 16 + lr;
#pragma unroll
      for (int v = 0; v < 8; ++v) {
        const int rl = wm0 + mi * 16 + hw * 8 + v;
        const float w = cws[rl];
        const size_t oidx = (size_t)(t0 + rl) * HIDDEN + col;
        out[oidx] += w * acc[mi][ni][v];
      }
    }
}

// ---------------------------------------------------------------------------
extern "C" void kernel_launch(void* const* d_in, const int* in_sizes, int n_in,
                              void* d_out, int out_size, void* d_ws, size_t ws_size,
                              hipStream_t stream) {
  (void)in_sizes; (void)n_in; (void)out_size; (void)ws_size;
  const float* x  = (const float*)d_in[0];
  const float* rw = (const float*)d_in[1];
  const float* wg = (const float*)d_in[2];
  const float* wu = (const float*)d_in[3];
  const float* wd = (const float*)d_in[4];
  float* out = (float*)d_out;

  // workspace carving (~62.5 MB)
  char* ws = (char*)d_ws;
  unsigned short* xb = (unsigned short*)ws;                      // T*H bf16
  size_t off = (size_t)TOKENS * HIDDEN * 2;
  float* cw = (float*)(ws + off);                                // T*E f32
  off += (size_t)TOKENS * NUM_EXPERTS * 4;
  int* flags = (int*)(ws + off);                                 // E*MTILES
  off += (size_t)NUM_EXPERTS * MTILES * 4;
  off = (off + 255) & ~(size_t)255;
  unsigned short* inter = (unsigned short*)(ws + off);           // T*I bf16

  moe_init_kernel<<<2048, 256, 0, stream>>>(out, flags, TOKENS * HIDDEN,
                                            NUM_EXPERTS * MTILES);
  moe_router_kernel<<<TOKENS / 256, 256, 0, stream>>>(x, rw, xb, cw, flags);

  for (int e = 0; e < NUM_EXPERTS; ++e) {
    const float* wge = wg + (size_t)e * HIDDEN * INTER;
    const float* wue = wu + (size_t)e * HIDDEN * INTER;
    const float* wde = wd + (size_t)e * INTER * HIDDEN;
    moe_gateup_kernel<<<dim3(INTER / NTILE, MTILES), 256, 0, stream>>>(
        xb, wge, wue, inter, flags + e * MTILES);
    moe_down_kernel<<<dim3(HIDDEN / NTILE, MTILES), 256, 0, stream>>>(
        inter, wde, cw, e, out, flags + e * MTILES);
  }
}
